// SparseMoE_36498632081499
// MI455X (gfx1250) — matrix operand order repaired and checked by reading, not compiled, not run on hardware
//
#include <hip/hip_runtime.h>
#include <math.h>

// Problem constants (match reference)
#define BSZ  2
#define SEQ  2048
#define DIM  1024          // D
#define NEXP 8             // E
#define HID  4096          // H
#define TOK  (BSZ*SEQ)     // 4096 tokens

typedef __attribute__((ext_vector_type(16))) _Float16 v16h;
typedef __attribute__((ext_vector_type(8)))  _Float16 v8h;
typedef __attribute__((ext_vector_type(8)))  float    v8f;

// LDS geometry for the FFN kernel (f16 elements; strides multiple of 8 so
// every WMMA A-fragment is a 16B-aligned ds_load_b128)
#define XS   1032          // x tile row stride (1024 + 8 pad)
#define HSR  136           // h tile row stride (128 + 8 pad)
#define NB   128           // H columns processed per outer iteration (8 waves x 16)

// A-operand fragment, contiguous k order chosen here (B loads use the same):
// lane<16: K = base..base+15 ; lanes 16-31: +16
__device__ __forceinline__ v16h ld_a(const _Float16* p) {
    v8h a = *(const v8h*)p;
    v8h b = *(const v8h*)(p + 8);
    return __builtin_shufflevector(a, b, 0,1,2,3,4,5,6,7,8,9,10,11,12,13,14,15);
}

// ------------------------------------------------------------------ init ----
__global__ __launch_bounds__(256) void moe_init(float* __restrict__ outF,
                                                int* __restrict__ counts) {
    long i = (long)blockIdx.x * 256 + threadIdx.x;
    if (i < (long)TOK * DIM) outF[i] = 0.f;
    if (i < NEXP) counts[i] = 0;
}

// --------------------------------------------------- weight cvt+transpose ----
// src: [E][R][C] f32  ->  dst: [E][C][R] f16   (64x64 tiles through LDS,
// coalesced on both the read (over C) and the write (over R) side)
__global__ __launch_bounds__(256) void moe_tconv(const float* __restrict__ src,
                                                 _Float16* __restrict__ dst,
                                                 int R, int Cc) {
    __shared__ _Float16 t[64][72];
    const int e  = blockIdx.z;
    const int r0 = blockIdx.y * 64, c0 = blockIdx.x * 64;
    const float*  s = src + (size_t)e * R * Cc;
    _Float16*     d = dst + (size_t)e * Cc * R;
    const int tid = threadIdx.x;
    for (int i = tid; i < 64 * 64; i += 256) {
        int rr = i >> 6, cc = i & 63;
        t[rr][cc] = (_Float16)s[(size_t)(r0 + rr) * Cc + (c0 + cc)];
    }
    __syncthreads();
    for (int i = tid; i < 64 * 64; i += 256) {
        int cc = i >> 6, rr = i & 63;
        d[(size_t)(c0 + cc) * R + (r0 + rr)] = t[rr][cc];
    }
}

// ---------------------------------------------------------------- router ----
// One 128-thread block per token: logits = x@Wr+br, noise_logits = x@Wn+bn,
// noisy = logits + noise*softplus(noise_logits), top-2, sparse softmax gates,
// dense softmax(logits) -> second output. Appends token to expert lists.
__global__ __launch_bounds__(128) void moe_router(
    const float* __restrict__ x,  const float* __restrict__ noise,
    const float* __restrict__ Wr, const float* __restrict__ br,
    const float* __restrict__ Wn, const float* __restrict__ bn,
    float* __restrict__ gates, float* __restrict__ smg,
    int* __restrict__ lists, int* __restrict__ counts)
{
    __shared__ float red[128 * 16];
    const int tok = blockIdx.x;
    const int tid = threadIdx.x;

    float aR[NEXP], aN[NEXP];
    #pragma unroll
    for (int e = 0; e < NEXP; ++e) { aR[e] = 0.f; aN[e] = 0.f; }

    const float* xr = x + (size_t)tok * DIM;
    for (int d = tid; d < DIM; d += 128) {
        float xv = xr[d];
        #pragma unroll
        for (int e = 0; e < NEXP; ++e) {
            aR[e] += xv * Wr[d * NEXP + e];
            aN[e] += xv * Wn[d * NEXP + e];
        }
    }
    #pragma unroll
    for (int e = 0; e < NEXP; ++e) {
        red[tid * 16 + e]     = aR[e];
        red[tid * 16 + 8 + e] = aN[e];
    }
    __syncthreads();
    if (tid < 16) {
        float s = 0.f;
        for (int t = 0; t < 128; ++t) s += red[t * 16 + tid];
        red[tid] = s;
    }
    __syncthreads();
    if (tid == 0) {
        float logits[NEXP], noisy[NEXP], t8[NEXP];
        #pragma unroll
        for (int e = 0; e < NEXP; ++e) logits[e] = red[e] + br[e];
        #pragma unroll
        for (int e = 0; e < NEXP; ++e) {
            float z  = red[8 + e] + bn[e];
            float sp = (z > 20.f) ? z : log1pf(expf(z));      // softplus
            noisy[e] = logits[e] + noise[(size_t)tok * NEXP + e] * sp;
        }
        int i1 = 0;
        for (int e = 1; e < NEXP; ++e) if (noisy[e] > noisy[i1]) i1 = e;
        int i2 = (i1 == 0) ? 1 : 0;
        for (int e = 0; e < NEXP; ++e) if (e != i1 && noisy[e] > noisy[i2]) i2 = e;
        float m  = fmaxf(noisy[i1], noisy[i2]);
        float g1 = expf(noisy[i1] - m), g2 = expf(noisy[i2] - m);
        float iv = 1.f / (g1 + g2);
        #pragma unroll
        for (int e = 0; e < NEXP; ++e) gates[(size_t)tok * NEXP + e] = 0.f;
        gates[(size_t)tok * NEXP + i1] = g1 * iv;
        gates[(size_t)tok * NEXP + i2] = g2 * iv;
        float mm = logits[0];
        #pragma unroll
        for (int e = 1; e < NEXP; ++e) mm = fmaxf(mm, logits[e]);
        float s = 0.f;
        #pragma unroll
        for (int e = 0; e < NEXP; ++e) { t8[e] = expf(logits[e] - mm); s += t8[e]; }
        float is = 1.f / s;
        #pragma unroll
        for (int e = 0; e < NEXP; ++e) smg[(size_t)tok * NEXP + e] = t8[e] * is;
        int p1 = atomicAdd(&counts[i1], 1); lists[i1 * TOK + p1] = tok;
        int p2 = atomicAdd(&counts[i2], 1); lists[i2 * TOK + p2] = tok;
    }
}

// ------------------------------------------------------------------- ffn ----
// blockIdx.x = expert, blockIdx.y = 16-token tile within that expert's list.
// 256 threads = 8 waves. Weights come pre-transposed in f16 (K contiguous),
// so every B fragment is two direct global_load_b128 — no LDS staging, no
// conversion in the hot loop. Only x (A, shared across waves) and the ReLU'd
// h chunk live in LDS.
__global__ __launch_bounds__(256) void moe_ffn(
    const float* __restrict__ x,
    const _Float16* __restrict__ W1T,   // [E][H][D] f16 (K=D contiguous)
    const float* __restrict__ b1,
    const _Float16* __restrict__ W2T,   // [E][D][H] f16 (K=H contiguous)
    const float* __restrict__ b2,
    const float* __restrict__ gates, const int* __restrict__ lists,
    const int* __restrict__ counts, float* __restrict__ outF)
{
    __shared__ _Float16 xs[16 * XS];    // x tile, f16
    __shared__ _Float16 hs[16 * HSR];   // relu(h) chunk
    __shared__ float    gate_s[16];
    __shared__ int      tok_s[16];

    const int e    = blockIdx.x;
    const int cnt  = counts[e];
    const int base = blockIdx.y * 16;
    if (base >= cnt) return;
    const int tid = threadIdx.x;

    if (tid < 16) {
        int r  = base + tid;
        int tk = (r < cnt) ? lists[e * TOK + r] : -1;
        tok_s[tid]  = tk;
        gate_s[tid] = (tk >= 0) ? gates[(size_t)tk * NEXP + e] : 0.f;
    }
    __syncthreads();
    for (int i = tid; i < 16 * DIM; i += 256) {     // gather rows, coalesced
        int row = i >> 10, col = i & (DIM - 1);
        int tk  = tok_s[row];
        xs[row * XS + col] = (_Float16)((tk >= 0) ? x[(size_t)tk * DIM + col] : 0.f);
    }
    __syncthreads();

    const int wv  = tid >> 5, lane = tid & 31;
    const int hlf = lane >> 4, l16 = lane & 15;
    const _Float16* W1Te = W1T + (size_t)e * HID * DIM;
    const _Float16* W2Te = W2T + (size_t)e * DIM * HID;

    v8f C[8];
    #pragma unroll
    for (int n = 0; n < 8; ++n)
        #pragma unroll
        for (int v = 0; v < 8; ++v) C[n][v] = 0.f;

    for (int hb = 0; hb < HID; hb += NB) {
        // ---- GEMM1: wave wv computes h[16, 16] at cols hb+wv*16.. over K=D.
        // B row for this lane: W1T[hcol][*] ; fragment = 32 contiguous bytes.
        const int hcol = hb + wv * 16 + l16;
        const _Float16* w1row = W1Te + (size_t)hcol * DIM + 16 * hlf;
        if (hb + NB < HID)                           // gfx1250 global_prefetch
            __builtin_prefetch(W1Te + (size_t)(hcol + NB) * DIM, 0, 0);

        v8f hacc;
        #pragma unroll
        for (int v = 0; v < 8; ++v) hacc[v] = 0.f;
        #pragma unroll 4
        for (int k0 = 0; k0 < DIM; k0 += 32) {
            v16h A  = ld_a(xs + l16 * XS + k0 + 16 * hlf);
            v16h Bm = *(const v16h*)(w1row + k0);    // 32B aligned
            hacc = __builtin_amdgcn_wmma_f32_16x16x32_f16(
                       false, A, false, Bm, (short)0, hacc, false, false);
        }
        // bias + ReLU; C-layout (M = v + 8*hlf, N = l16) -> hs[token][h]
        float b1v = b1[e * HID + hcol];
        #pragma unroll
        for (int v = 0; v < 8; ++v) {
            float hv = fmaxf(hacc[v] + b1v, 0.f);
            hs[(v + 8 * hlf) * HSR + wv * 16 + l16] = (_Float16)hv;
        }
        __syncthreads();                             // hs ready for all waves

        // ---- GEMM2: y[16, wave's 128 cols] += h[16,NB] @ W2[hb:hb+NB, cols]
        #pragma unroll
        for (int kk = 0; kk < NB; kk += 32) {
            v16h A2 = ld_a(hs + l16 * HSR + kk + 16 * hlf);
            #pragma unroll
            for (int n = 0; n < 8; ++n) {
                int dc = wv * 128 + n * 16 + l16;
                v16h B2 = *(const v16h*)(W2Te + (size_t)dc * HID + hb + kk + 16 * hlf);
                C[n] = __builtin_amdgcn_wmma_f32_16x16x32_f16(
                           false, A2, false, B2, (short)0, C[n], false, false);
            }
        }
        __syncthreads();                             // hs consumed; safe to rewrite
    }

    // epilogue: final[tok] += gate * (y + b2)  (the two selected gates sum to
    // 1, so b2 contributes exactly once, matching the reference)
    float gv[8]; int tk[8];
    #pragma unroll
    for (int v = 0; v < 8; ++v) { int r = v + 8 * hlf; gv[v] = gate_s[r]; tk[v] = tok_s[r]; }
    #pragma unroll
    for (int n = 0; n < 8; ++n) {
        int dc = wv * 128 + n * 16 + l16;
        float b2v = b2[e * DIM + dc];
        #pragma unroll
        for (int v = 0; v < 8; ++v) {
            if (tk[v] >= 0) {
                __hip_atomic_fetch_add(&outF[(size_t)tk[v] * DIM + dc],
                                       gv[v] * (C[n][v] + b2v),
                                       __ATOMIC_RELAXED, __HIP_MEMORY_SCOPE_AGENT);
            }
        }
    }
}

// ---------------------------------------------------------------- launch ----
extern "C" void kernel_launch(void* const* d_in, const int* in_sizes, int n_in,
                              void* d_out, int out_size, void* d_ws, size_t ws_size,
                              hipStream_t stream) {
    const float* x     = (const float*)d_in[0];
    const float* noise = (const float*)d_in[1];
    const float* Wr    = (const float*)d_in[2];
    const float* br    = (const float*)d_in[3];
    const float* Wn    = (const float*)d_in[4];
    const float* bn    = (const float*)d_in[5];
    const float* W1    = (const float*)d_in[6];
    const float* b1    = (const float*)d_in[7];
    const float* W2    = (const float*)d_in[8];
    const float* b2    = (const float*)d_in[9];

    float* outF = (float*)d_out;                       // [B,S,D] final
    float* smg  = outF + (size_t)TOK * DIM;            // [B,S,E] softmax gating

    // workspace layout (~134.5 MB): control + f16 transposed weights
    char*  ws     = (char*)d_ws;
    int*   counts = (int*)ws;                                        // [E]
    float* gates  = (float*)(ws + 256);                              // [TOK,E]
    int*   lists  = (int*)(ws + 256 + (size_t)TOK * NEXP * 4);       // [E,TOK]
    size_t wOff   = 256 + (size_t)TOK * NEXP * 4 * 2;                // = 262400
    _Float16* W1T = (_Float16*)(ws + wOff);                          // [E][H][D]
    _Float16* W2T = W1T + (size_t)NEXP * HID * DIM;                  // [E][D][H]

    moe_init<<<(TOK * DIM + 255) / 256, 256, 0, stream>>>(outF, counts);
    // one-time (per launch) weight convert+transpose to K-contiguous f16
    moe_tconv<<<dim3(HID / 64, DIM / 64, NEXP), 256, 0, stream>>>(W1, W1T, DIM, HID);
    moe_tconv<<<dim3(DIM / 64, HID / 64, NEXP), 256, 0, stream>>>(W2, W2T, HID, DIM);
    moe_router<<<TOK, 128, 0, stream>>>(x, noise, Wr, br, Wn, bn,
                                        gates, smg, lists, counts);
    dim3 grid(NEXP, TOK / 16);                         // covers worst-case skew
    moe_ffn<<<grid, 256, 0, stream>>>(x, W1T, b1, W2T, b2,
                                      gates, lists, counts, outF);
}
